// PromptAdapater_24790551233039
// MI455X (gfx1250) — matrix operand order repaired and checked by reading, not compiled, not run on hardware
//
#include <hip/hip_runtime.h>
#include <hip/hip_bf16.h>
#include <math.h>

#define B_    16
#define NQ    8
#define NK    4096
#define C_    256
#define HEADS 8
#define CI    128
#define HD    16
#define NSP   4
#define F32_TINY 1.17549435e-38f
#define TWO_PI   6.28318530717958647692f

typedef _Float16 half_t;
typedef __attribute__((ext_vector_type(16))) _Float16 v16h;
typedef __attribute__((ext_vector_type(8)))  float    v8f;

// ---------------------------------------------------------------------------
// Kernel 1: g[b,k,c] = guiding[b,c,k] * keys[b,k,c]   (LDS transpose of guiding)
// ---------------------------------------------------------------------------
__global__ __launch_bounds__(256) void k_g_compute(const float* __restrict__ keys,
                                                   const float* __restrict__ guiding,
                                                   float* __restrict__ g) {
  int b  = blockIdx.y;
  int k0 = blockIdx.x * 64;
  __shared__ float lt[32][65];
  const int tid = threadIdx.x;
  for (int cc = 0; cc < C_ / 32; ++cc) {
    for (int ii = 0; ii < 8; ++ii) {
      int i = tid + 256 * ii;          // 2048 elements
      int cl = i >> 6, kl = i & 63;
      lt[cl][kl] = guiding[((size_t)b * C_ + cc * 32 + cl) * NK + k0 + kl];
    }
    __syncthreads();
    for (int ii = 0; ii < 8; ++ii) {
      int i = tid + 256 * ii;
      int kl = i >> 5, cl = i & 31;
      size_t idx = ((size_t)(b * NK + k0 + kl)) * C_ + cc * 32 + cl;
      g[idx] = keys[idx] * lt[cl][kl];
    }
    __syncthreads();
  }
}

// ---------------------------------------------------------------------------
// Kernel 2: WMMA projections. kh = (g+key_pe)@Wk + bk ; vh = g@Wv + bv
// Output layout: kh/vh[b][h][k][d]  (f16)
// Workgroup: 256 thr (8 waves), tile = 128 rows x 128 cols, K chunks of 32.
// ---------------------------------------------------------------------------
__global__ __launch_bounds__(256) void k_proj(const float* __restrict__ g,
                                              const float* __restrict__ key_pe,
                                              const float* __restrict__ Wk,
                                              const float* __restrict__ bk,
                                              const float* __restrict__ Wv,
                                              const float* __restrict__ bv,
                                              half_t* __restrict__ kh,
                                              half_t* __restrict__ vh) {
  int b  = blockIdx.y;
  int k0 = blockIdx.x * 128;
  __shared__ half_t sAk[128 * 32];   // kin = g + key_pe, [row][K]
  __shared__ half_t sAg[128 * 32];   // g,               [row][K]
  __shared__ half_t sBk[128 * 32];   // Wk chunk transposed, [col][K]
  __shared__ half_t sBv[128 * 32];
  const int tid  = threadIdx.x;
  const int lane = tid & 31;
  const int wave = tid >> 5;
  const int m    = lane & 15;
  const int hi   = lane >> 4;

  v8f acck[8], accv[8];
  for (int n = 0; n < 8; ++n)
    for (int r = 0; r < 8; ++r) { acck[n][r] = 0.f; accv[n][r] = 0.f; }

  for (int kc = 0; kc < C_ / 32; ++kc) {
    for (int ii = 0; ii < 16; ++ii) {
      int i = tid + 256 * ii;          // 4096
      int r = i >> 5, kk = i & 31;
      size_t gi = ((size_t)(b * NK + k0 + r)) * C_ + kc * 32 + kk;
      float gv = g[gi];
      sAg[i] = (half_t)gv;
      sAk[i] = (half_t)(gv + key_pe[gi]);
    }
    for (int ii = 0; ii < 16; ++ii) {
      int i = tid + 256 * ii;          // 4096
      int col = i & 127, kk = i >> 7;
      sBk[col * 32 + kk] = (half_t)Wk[(size_t)(kc * 32 + kk) * CI + col];
      sBv[col * 32 + kk] = (half_t)Wv[(size_t)(kc * 32 + kk) * CI + col];
    }
    __syncthreads();
    // A fragments: row = lane%16, K = (lane/16)*16 + j
    v16h ak, ag;
    int rowoff = (wave * 16 + m) * 32;
    for (int j = 0; j < 16; ++j) {
      int K = hi * 16 + j;
      ak[j] = sAk[rowoff + K];
      ag[j] = sAg[rowoff + K];
    }
    for (int n = 0; n < 8; ++n) {
      // B fragment: col = n*16 + lane%16, K = (lane/16)*16 + j (contiguous in LDS)
      v16h bkf, bvf;
      int cb = (n * 16 + m) * 32 + hi * 16;
      for (int j = 0; j < 16; ++j) { bkf[j] = sBk[cb + j]; bvf[j] = sBv[cb + j]; }
      acck[n] = __builtin_amdgcn_wmma_f32_16x16x32_f16(false, ak, false, bkf,
                                                       (short)0, acck[n], false, false);
      accv[n] = __builtin_amdgcn_wmma_f32_16x16x32_f16(false, ag, false, bvf,
                                                       (short)0, accv[n], false, false);
    }
    __syncthreads();
  }
  // C/D layout: col = lane%16 (-> d), row = r + 8*(lane/16); head = n (16-col block)
  for (int n = 0; n < 8; ++n) {
    float bkv = bk[n * 16 + m];
    float bvv = bv[n * 16 + m];
    for (int r = 0; r < 8; ++r) {
      int k = k0 + wave * 16 + r + 8 * hi;
      size_t o = (((size_t)b * HEADS + n) * NK + k) * HD + m;
      kh[o] = (half_t)(acck[n][r] + bkv);
      vh[o] = (half_t)(accv[n][r] + bvv);
    }
  }
}

// ---------------------------------------------------------------------------
// Kernel 3: qh = (queries + query_pe) @ Wq + bq   (tiny: 128 rows total)
// ---------------------------------------------------------------------------
__global__ __launch_bounds__(128) void k_qh(const float* __restrict__ queries,
                                            const float* __restrict__ query_pe,
                                            const float* __restrict__ Wq,
                                            const float* __restrict__ bq,
                                            float* __restrict__ qh) {
  int b = blockIdx.x, ci = threadIdx.x;
  __shared__ float xin[NQ][C_];
  for (int i = threadIdx.x; i < NQ * C_; i += 128) {
    int q = i >> 8, c = i & 255;
    size_t idx = ((size_t)(b * NQ + q)) * C_ + c;
    xin[q][c] = queries[idx] + query_pe[idx];
  }
  __syncthreads();
  for (int q = 0; q < NQ; ++q) {
    float acc = bq[ci];
    for (int c = 0; c < C_; ++c) acc += xin[q][c] * Wq[(size_t)c * CI + ci];
    qh[((size_t)(b * NQ + q)) * CI + ci] = acc;
  }
}

// ---------------------------------------------------------------------------
// Kernel 4: attention per (b,h). Two-pass softmax over NK, LDS tree reductions.
// ---------------------------------------------------------------------------
__global__ __launch_bounds__(256) void k_attn(const float* __restrict__ qh,
                                              const half_t* __restrict__ kh,
                                              const half_t* __restrict__ vh,
                                              float* __restrict__ attn_out) {
  int h = blockIdx.x, b = blockIdx.y;
  int tid = threadIdx.x;
  __shared__ float qs[NQ][HD];
  __shared__ float red[256];
  __shared__ float pacc[HD * 256];
  for (int i = tid; i < NQ * HD; i += 256) {
    int q = i / HD, d = i % HD;
    qs[q][d] = qh[((size_t)(b * NQ + q)) * CI + h * HD + d];
  }
  __syncthreads();
  const half_t* khb = kh + ((size_t)b * HEADS + h) * NK * HD;
  const half_t* vhb = vh + ((size_t)b * HEADS + h) * NK * HD;
  for (int q = 0; q < NQ; ++q) {
    float mloc = -INFINITY;
    for (int j = 0; j < NK / 256; ++j) {
      int k = tid + 256 * j;
      const half_t* kp = khb + (size_t)k * HD;
      float s = 0.f;
      for (int d = 0; d < HD; ++d) s += qs[q][d] * (float)kp[d];
      mloc = fmaxf(mloc, s * 0.25f);
    }
    red[tid] = mloc; __syncthreads();
    for (int off = 128; off > 0; off >>= 1) {
      if (tid < off) red[tid] = fmaxf(red[tid], red[tid + off]);
      __syncthreads();
    }
    float M = red[0]; __syncthreads();
    float ssum = 0.f, oacc[HD];
    for (int d = 0; d < HD; ++d) oacc[d] = 0.f;
    for (int j = 0; j < NK / 256; ++j) {
      int k = tid + 256 * j;
      const half_t* kp = khb + (size_t)k * HD;
      float s = 0.f;
      for (int d = 0; d < HD; ++d) s += qs[q][d] * (float)kp[d];
      float e = expf(s * 0.25f - M);
      ssum += e;
      const half_t* vp = vhb + (size_t)k * HD;
      for (int d = 0; d < HD; ++d) oacc[d] += e * (float)vp[d];
    }
    red[tid] = ssum; __syncthreads();
    for (int off = 128; off > 0; off >>= 1) {
      if (tid < off) red[tid] += red[tid + off];
      __syncthreads();
    }
    float S = red[0]; __syncthreads();
    for (int d = 0; d < HD; ++d) pacc[d * 256 + tid] = oacc[d];
    __syncthreads();
    for (int off = 128; off > 0; off >>= 1) {
      if (tid < off)
        for (int d = 0; d < HD; ++d) pacc[d * 256 + tid] += pacc[d * 256 + tid + off];
      __syncthreads();
    }
    if (tid < HD)
      attn_out[((size_t)(b * NQ + q)) * CI + h * HD + tid] = pacc[tid * 256] / S;
    __syncthreads();
  }
}

// ---------------------------------------------------------------------------
// Kernel 5: q_final = attn_out @ Wo + bo ; unc_tok / ref_tok MLPs (per batch)
// ---------------------------------------------------------------------------
__global__ __launch_bounds__(256) void k_post(const float* __restrict__ attn_out,
    const float* __restrict__ Wo,  const float* __restrict__ bo,
    const float* __restrict__ su,  const float* __restrict__ sr,
    const float* __restrict__ uW1, const float* __restrict__ ub1,
    const float* __restrict__ uW2, const float* __restrict__ ub2,
    const float* __restrict__ uW3, const float* __restrict__ ub3,
    const float* __restrict__ rW1, const float* __restrict__ rb1,
    const float* __restrict__ rW2, const float* __restrict__ rb2,
    const float* __restrict__ rW3, const float* __restrict__ rb3,
    float* __restrict__ q_final, float* __restrict__ q1_ws,
    float* __restrict__ unc_tok, float* __restrict__ ref_tok) {
  int b = blockIdx.x, tid = threadIdx.x;
  __shared__ float ao[NQ * CI];
  __shared__ float q1s[C_];
  __shared__ float h1u[64], h1r[64], h2u[64], h2r[64];
  for (int i = tid; i < NQ * CI; i += 256) ao[i] = attn_out[(size_t)b * NQ * CI + i];
  __syncthreads();
  int c = tid;
  for (int q = 0; q < NQ; ++q) {
    float acc = bo[c];
    for (int i = 0; i < CI; ++i) acc += ao[q * CI + i] * Wo[(size_t)i * C_ + c];
    q_final[((size_t)(b * NQ + q)) * C_ + c] = acc;
    if (q == 1) { q1s[c] = acc; q1_ws[(size_t)b * C_ + c] = acc; }
  }
  __syncthreads();
  if (tid < 64) {
    float au = ub1[tid], ar = rb1[tid];
    for (int i = 0; i < C_; ++i) {
      au += su[i] * uW1[(size_t)i * 64 + tid];
      ar += sr[i] * rW1[(size_t)i * 64 + tid];
    }
    for (int i = 0; i < C_; ++i) {
      float q1v = q1s[i];
      au += q1v * uW1[(size_t)(C_ + i) * 64 + tid];
      ar += q1v * rW1[(size_t)(C_ + i) * 64 + tid];
    }
    h1u[tid] = fmaxf(au, 0.f);
    h1r[tid] = fmaxf(ar, 0.f);
  }
  __syncthreads();
  if (tid < 64) {
    float au = ub2[tid], ar = rb2[tid];
    for (int i = 0; i < 64; ++i) {
      au += h1u[i] * uW2[(size_t)i * 64 + tid];
      ar += h1r[i] * rW2[(size_t)i * 64 + tid];
    }
    h2u[tid] = fmaxf(au, 0.f);
    h2r[tid] = fmaxf(ar, 0.f);
  }
  __syncthreads();
  {
    float au = ub3[c], ar = rb3[c];
    for (int i = 0; i < 64; ++i) {
      au += h2u[i] * uW3[(size_t)i * C_ + c];
      ar += h2r[i] * rW3[(size_t)i * C_ + c];
    }
    unc_tok[(size_t)b * C_ + c] = au;
    ref_tok[(size_t)b * C_ + c] = ar;
  }
}

// ---------------------------------------------------------------------------
// Kernel 6: ref_map = sigmoid(unc·g) * (sigmoid(ref·g) - sigmoid(q1·g))
// LDS-tiled over g; 2 threads per n (split the C reduction), 128 n per block.
// ---------------------------------------------------------------------------
__global__ __launch_bounds__(256) void k_masks(const float* __restrict__ g,
                                               const float* __restrict__ unc_tok,
                                               const float* __restrict__ ref_tok,
                                               const float* __restrict__ q1_ws,
                                               float* __restrict__ ref_map) {
  int b  = blockIdx.y;
  int n0 = blockIdx.x * 128;
  int tid = threadIdx.x;
  __shared__ float ut[C_], rt[C_], qt[C_];
  __shared__ float sg[128 * 33];
  __shared__ float comb[256 * 3];
  ut[tid] = unc_tok[(size_t)b * C_ + tid];
  rt[tid] = ref_tok[(size_t)b * C_ + tid];
  qt[tid] = q1_ws[(size_t)b * C_ + tid];
  int nl = tid >> 1, half = tid & 1;
  float aU = 0.f, aR = 0.f, aC = 0.f;
  for (int cc = 0; cc < C_ / 32; ++cc) {
    __syncthreads();
    for (int ii = 0; ii < 16; ++ii) {
      int i = tid + 256 * ii;          // 4096
      int r = i >> 5, cl = i & 31;
      sg[r * 33 + cl] = g[((size_t)(b * NK + n0 + r)) * C_ + cc * 32 + cl];
    }
    __syncthreads();
    for (int kk = 0; kk < 16; ++kk) {
      int cl = half * 16 + kk;
      float gv = sg[nl * 33 + cl];
      int cg = cc * 32 + cl;
      aU += ut[cg] * gv;
      aR += rt[cg] * gv;
      aC += qt[cg] * gv;
    }
  }
  comb[tid] = aU; comb[256 + tid] = aR; comb[512 + tid] = aC;
  __syncthreads();
  if (half == 0) {
    aU += comb[tid + 1]; aR += comb[256 + tid + 1]; aC += comb[512 + tid + 1];
    float un = 1.f / (1.f + expf(-aU));
    float sR = 1.f / (1.f + expf(-aR));
    float sC = 1.f / (1.f + expf(-aC));
    ref_map[(size_t)b * NK + n0 + nl] = un * (sR - sC);
  }
}

// ---------------------------------------------------------------------------
// Kernel 7: gumbel soft top-k (4 softmax rounds) + hard top-4 + index sort
// grid (sign, b); each thread owns 16 strided elements.
// ---------------------------------------------------------------------------
__global__ __launch_bounds__(256) void k_topk(const float* __restrict__ ref_map,
                                              const float* __restrict__ gumbel_p,
                                              const float* __restrict__ gumbel_n,
                                              int* __restrict__ topk_idx,
                                              float* __restrict__ topk_ret) {
  int sgn = blockIdx.x, b = blockIdx.y;
  int tid = threadIdx.x;
  __shared__ float red[256];
  __shared__ int   redi[256];
  float gval[16], onehot[16], khot[16];
  for (int j = 0; j < 16; ++j) {
    int k = tid + 256 * j;
    float s = ref_map[(size_t)b * NK + k];
    if (sgn) s = -s;
    float gm = sgn ? gumbel_n[(size_t)b * NK + k] : gumbel_p[(size_t)b * NK + k];
    gval[j] = s + gm;
    onehot[j] = 0.f; khot[j] = 0.f;
  }
  for (int it = 0; it < NSP; ++it) {
    float mloc = -INFINITY;
    for (int j = 0; j < 16; ++j) {
      gval[j] += logf(fmaxf(1.f - onehot[j], F32_TINY));
      mloc = fmaxf(mloc, gval[j]);
    }
    red[tid] = mloc; __syncthreads();
    for (int off = 128; off > 0; off >>= 1) {
      if (tid < off) red[tid] = fmaxf(red[tid], red[tid + off]);
      __syncthreads();
    }
    float M = red[0]; __syncthreads();
    float e[16], ssum = 0.f;
    for (int j = 0; j < 16; ++j) { e[j] = expf(gval[j] - M); ssum += e[j]; }
    red[tid] = ssum; __syncthreads();
    for (int off = 128; off > 0; off >>= 1) {
      if (tid < off) red[tid] += red[tid + off];
      __syncthreads();
    }
    float S = red[0]; __syncthreads();
    for (int j = 0; j < 16; ++j) { onehot[j] = e[j] / S; khot[j] += onehot[j]; }
  }
  float kv[16];
  for (int j = 0; j < 16; ++j) kv[j] = khot[j];
  int sel_i[NSP]; float sel_v[NSP];
  for (int r = 0; r < NSP; ++r) {
    float bv = -INFINITY; int bi = NK;
    for (int j = 0; j < 16; ++j) {
      int k = tid + 256 * j;
      if (kv[j] > bv || (kv[j] == bv && k < bi)) { bv = kv[j]; bi = k; }
    }
    red[tid] = bv; redi[tid] = bi; __syncthreads();
    for (int off = 128; off > 0; off >>= 1) {
      if (tid < off) {
        if (red[tid + off] > red[tid] ||
            (red[tid + off] == red[tid] && redi[tid + off] < redi[tid])) {
          red[tid] = red[tid + off]; redi[tid] = redi[tid + off];
        }
      }
      __syncthreads();
    }
    int win = redi[0]; float wv = red[0];
    __syncthreads();
    sel_i[r] = win; sel_v[r] = wv;
    if ((win & 255) == tid) kv[win >> 8] = -INFINITY;
  }
  if (tid == 0) {
    for (int a = 0; a < NSP; ++a)
      for (int bb = a + 1; bb < NSP; ++bb)
        if (sel_i[bb] < sel_i[a]) {
          int ti = sel_i[a]; sel_i[a] = sel_i[bb]; sel_i[bb] = ti;
          float tv = sel_v[a]; sel_v[a] = sel_v[bb]; sel_v[bb] = tv;
        }
    for (int r = 0; r < NSP; ++r) {
      topk_idx[((size_t)b * 2 + sgn) * NSP + r] = sel_i[r];
      topk_ret[((size_t)b * 2 + sgn) * NSP + r] = (1.f - sel_v[r]) + sel_v[r];
    }
  }
}

// ---------------------------------------------------------------------------
// Kernel 8: assemble output (B, 32, C): queries rows 0..15, pe rows 16..31.
// ---------------------------------------------------------------------------
__global__ __launch_bounds__(256) void k_out(const float* __restrict__ q_final,
    const float* __restrict__ ref_tok,
    const float* __restrict__ keys, const float* __restrict__ g,
    const float* __restrict__ query_pe, const float* __restrict__ pe_gauss,
    const float* __restrict__ point_pos, const float* __restrict__ point_neg,
    const int* __restrict__ topk_idx, const float* __restrict__ topk_ret,
    float* __restrict__ out) {
  int b = blockIdx.x, c = threadIdx.x;
  float* ob = out + (size_t)b * 32 * C_;
  for (int q = 0; q < NQ; ++q) {
    float v = (q == 1) ? ref_tok[(size_t)b * C_ + c]
                       : q_final[((size_t)(b * NQ + q)) * C_ + c];
    ob[q * C_ + c] = v;
  }
  int jj = c & 127;
  float pg0 = pe_gauss[jj], pg1 = pe_gauss[128 + jj];
  for (int sgn = 0; sgn < 2; ++sgn) {
    const float* pp = sgn ? point_neg : point_pos;
    for (int r = 0; r < NSP; ++r) {
      int n = topk_idx[((size_t)b * 2 + sgn) * NSP + r];
      float rv = topk_ret[((size_t)b * 2 + sgn) * NSP + r];
      size_t gi = ((size_t)(b * NK + n)) * C_ + c;
      ob[(8 + sgn * NSP + r) * C_ + c] = (keys[gi] + g[gi]) * rv;
      int x = n & 63, y = n >> 6;
      float fx = 2.f * ((x + 0.5f) / 64.f) - 1.f;
      float fy = 2.f * ((y + 0.5f) / 64.f) - 1.f;
      float proj = (fx * pg0 + fy * pg1) * TWO_PI;
      float pev = (c < 128) ? sinf(proj) : cosf(proj);
      ob[(24 + sgn * NSP + r) * C_ + c] = pev * rv + pp[c];
    }
  }
  for (int q = 0; q < NQ; ++q)
    ob[(16 + q) * C_ + c] = query_pe[((size_t)(b * NQ + q)) * C_ + c];
}

// ---------------------------------------------------------------------------
extern "C" void kernel_launch(void* const* d_in, const int* in_sizes, int n_in,
                              void* d_out, int out_size, void* d_ws, size_t ws_size,
                              hipStream_t stream) {
  (void)in_sizes; (void)n_in; (void)out_size; (void)ws_size;
  const float* queries  = (const float*)d_in[0];
  const float* keys     = (const float*)d_in[1];
  const float* query_pe = (const float*)d_in[2];
  const float* key_pe   = (const float*)d_in[3];
  const float* guiding  = (const float*)d_in[4];
  const float* gumbel_p = (const float*)d_in[5];
  const float* gumbel_n = (const float*)d_in[6];
  const float* su = (const float*)d_in[7];
  const float* sr = (const float*)d_in[8];
  const float* Wq = (const float*)d_in[9];  const float* bq = (const float*)d_in[10];
  const float* Wk = (const float*)d_in[11]; const float* bk = (const float*)d_in[12];
  const float* Wv = (const float*)d_in[13]; const float* bv = (const float*)d_in[14];
  const float* Wo = (const float*)d_in[15]; const float* bo = (const float*)d_in[16];
  const float* uW1 = (const float*)d_in[17]; const float* ub1 = (const float*)d_in[18];
  const float* uW2 = (const float*)d_in[19]; const float* ub2 = (const float*)d_in[20];
  const float* uW3 = (const float*)d_in[21]; const float* ub3 = (const float*)d_in[22];
  const float* rW1 = (const float*)d_in[23]; const float* rb1 = (const float*)d_in[24];
  const float* rW2 = (const float*)d_in[25]; const float* rb2 = (const float*)d_in[26];
  const float* rW3 = (const float*)d_in[27]; const float* rb3 = (const float*)d_in[28];
  const float* pe_gauss  = (const float*)d_in[29];
  const float* point_pos = (const float*)d_in[30];
  const float* point_neg = (const float*)d_in[31];

  char* ws = (char*)d_ws;
  size_t off = 0;
  auto alloc = [&](size_t bytes) -> void* {
    void* p = ws + off;
    off = (off + bytes + 255) & ~(size_t)255;
    return p;
  };
  float*  g_ws  = (float*) alloc((size_t)B_ * NK * C_ * 4);
  half_t* kh_ws = (half_t*)alloc((size_t)B_ * HEADS * NK * HD * 2);
  half_t* vh_ws = (half_t*)alloc((size_t)B_ * HEADS * NK * HD * 2);
  float*  qh_ws = (float*) alloc((size_t)B_ * NQ * CI * 4);
  float*  ao_ws = (float*) alloc((size_t)B_ * NQ * CI * 4);
  float*  qf_ws = (float*) alloc((size_t)B_ * NQ * C_ * 4);
  float*  q1_ws = (float*) alloc((size_t)B_ * C_ * 4);
  float*  ut_ws = (float*) alloc((size_t)B_ * C_ * 4);
  float*  rt_ws = (float*) alloc((size_t)B_ * C_ * 4);
  float*  rm_ws = (float*) alloc((size_t)B_ * NK * 4);
  int*    ti_ws = (int*)   alloc((size_t)B_ * 2 * NSP * 4);
  float*  tr_ws = (float*) alloc((size_t)B_ * 2 * NSP * 4);

  k_g_compute<<<dim3(NK / 64, B_), 256, 0, stream>>>(keys, guiding, g_ws);
  k_proj<<<dim3(NK / 128, B_), 256, 0, stream>>>(g_ws, key_pe, Wk, bk, Wv, bv, kh_ws, vh_ws);
  k_qh<<<dim3(B_), 128, 0, stream>>>(queries, query_pe, Wq, bq, qh_ws);
  k_attn<<<dim3(HEADS, B_), 256, 0, stream>>>(qh_ws, kh_ws, vh_ws, ao_ws);
  k_post<<<dim3(B_), 256, 0, stream>>>(ao_ws, Wo, bo, su, sr,
                                       uW1, ub1, uW2, ub2, uW3, ub3,
                                       rW1, rb1, rW2, rb2, rW3, rb3,
                                       qf_ws, q1_ws, ut_ws, rt_ws);
  k_masks<<<dim3(NK / 128, B_), 256, 0, stream>>>(g_ws, ut_ws, rt_ws, q1_ws, rm_ws);
  k_topk<<<dim3(2, B_), 256, 0, stream>>>(rm_ws, gumbel_p, gumbel_n, ti_ws, tr_ws);
  k_out<<<dim3(B_), 256, 0, stream>>>(qf_ws, rt_ws, keys, g_ws, query_pe, pe_gauss,
                                      point_pos, point_neg, ti_ws, tr_ws, (float*)d_out);
}
